// S_Module_43739946942675
// MI455X (gfx1250) — matrix operand order repaired and checked by reading, not compiled, not run on hardware
//
#include <hip/hip_runtime.h>
#include <hip/hip_bf16.h>
#include <math.h>

// ---------------- problem constants ----------------
constexpr int BATCH = 8;
constexpr int CH    = 256;
constexpr int HH    = 128;
constexpr int WW    = 128;
constexpr int HWSZ  = HH * WW;      // 16384
constexpr int C8    = 32;           // q/k channels
constexpr int CSE   = 16;           // SE hidden
constexpr float BN_EPS = 1e-5f;
constexpr float LN_EPS = 1e-5f;

// ---------------- WMMA / TDM types ----------------
typedef __attribute__((ext_vector_type(16))) __bf16    v16bf;
typedef __attribute__((ext_vector_type(8)))  float     v8f;
typedef __attribute__((ext_vector_type(4)))  unsigned  v4u;
typedef __attribute__((ext_vector_type(8)))  int       v8i;
typedef __attribute__((ext_vector_type(4)))  int       v4i;

__device__ inline v8f wmma_bf16(v16bf a, v16bf b, v8f c) {
    return __builtin_amdgcn_wmma_f32_16x16x32_bf16(false, a, false, b,
                                                   (short)0, c, false, false);
}

// ---- fragment-major LDS staging (inverse of ISA 7.12.2 layouts) ----
// A (16M x 32K bf16): lanes 0-15 hold M=lane with K {0..7,16..23}; lanes 16-31 hold K {8..15,24..31}.
__device__ inline void store_a_elem(v16bf* AF, int m, int k, float val) {
    const int hi   = (k >> 3) & 1;
    const int lane = m + (hi << 4);
    const int k2   = k - (hi << 3);
    const int i    = (k2 < 8) ? k2 : (k2 - 8);
    ((__bf16*)&AF[lane])[i] = (__bf16)val;
}
// B (32K x 16N bf16): lanes 0-15: N=lane, K {0..7,16..23}; lanes 16-31: K {8..15,24..31}.
__device__ inline void store_b_elem(v16bf* BF, int k, int n, float val) {
    const int lane = (((k >> 3) & 1) << 4) | n;
    ((__bf16*)&BF[lane])[(k & 7) | ((k >> 4) << 3)] = (__bf16)val;
}

// ---- TDM: async 2D tile load (rows x 32 contiguous fp32) into LDS ----
// D# per ISA ch.8: group0 {count=1, lds_addr, global_addr[56:0], type=2},
// group1 {data_size=4B, tensor_dim0=32, tensor_dim1=nrows, tile 32 x nrows, dim0_stride}.
__device__ inline void tdm_load_tile_f32(const float* gptr, void* ldsDst,
                                         unsigned rowStrideElems, unsigned nrows) {
    const unsigned long long ga = (unsigned long long)(const void*)gptr;
    const unsigned ldsAddr = (unsigned)(size_t)ldsDst;   // LDS aperture: low 32 bits
    const v4u g0 = { 1u,                                  // count=1
                     ldsAddr,
                     (unsigned)(ga & 0xFFFFFFFFu),
                     (unsigned)(((ga >> 32) & 0x01FFFFFFu) | (2u << 30)) }; // type=2
    const v8i g1 = { (int)(2u << 16),          // data_size = 4 bytes
                     (int)(32u << 16),         // tensor_dim0 = 32
                     (int)(nrows << 16),       // tensor_dim1
                     (int)(32u << 16),         // tile_dim0 = 32
                     (int)nrows,               // tile_dim1
                     (int)rowStrideElems,      // tensor_dim0_stride[31:0]
                     0, 0 };
    const v4i gz = { 0, 0, 0, 0 };
#if __has_include(<hip/amd_detail/amd_gfx1250_TDM.h>)
    const v8i gz8 = { 0, 0, 0, 0, 0, 0, 0, 0 };
    __builtin_amdgcn_tensor_load_to_lds(g0, g1, gz, gz, gz8, 0);
#else
    __builtin_amdgcn_tensor_load_to_lds(g0, g1, gz, gz, 0);
#endif
    __builtin_amdgcn_s_wait_tensorcnt(0);
}

// ============ K1: x + sincos pos-embed, per-(b,c) mean ============
__global__ void k_posembed_mean(const float* __restrict__ x,
                                float* __restrict__ xpe,
                                float* __restrict__ meanbc) {
    const int bc = blockIdx.x;            // b*CH + c
    const int c  = bc & (CH - 1);
    const float* xin  = x   + (size_t)bc * HWSZ;
    float*       xout = xpe + (size_t)bc * HWSZ;

    const int  dim   = CH / 2;
    const bool isW   = (c >= dim);
    const int  cc    = isW ? c - dim : c;
    const int  j     = cc >> 1;
    const bool isCos = (cc & 1);
    const float freq = expf(-(float)(2 * j) * (logf(10000.f) / (float)dim));

    float lsum = 0.f;
    for (int idx = threadIdx.x; idx < HWSZ; idx += blockDim.x) {
        const int h = idx >> 7, w = idx & (WW - 1);
        const float ang = (isW ? (float)w : (float)h) * freq;
        const float pe  = isCos ? cosf(ang) : sinf(ang);
        const float val = xin[idx] + pe;
        xout[idx] = val;
        lsum += val;
    }
    __shared__ float red[256];
    red[threadIdx.x] = lsum;
    __syncthreads();
    for (int s = 128; s > 0; s >>= 1) {
        if (threadIdx.x < s) red[threadIdx.x] += red[threadIdx.x + s];
        __syncthreads();
    }
    if (threadIdx.x == 0) meanbc[bc] = red[0] * (1.f / (float)HWSZ);
}

// ============ K2: SE MLP (tiny, one block) ============
__global__ void k_se(const float* __restrict__ meanbc,
                     const float* __restrict__ w1,
                     const float* __restrict__ w2,
                     float* __restrict__ sescale) {
    __shared__ float h1[BATCH][CSE];
    const int t = threadIdx.x;
    if (t < BATCH * CSE) {
        const int b = t / CSE, jj = t % CSE;
        float acc = 0.f;
        for (int c = 0; c < CH; ++c) acc += meanbc[b * CH + c] * w1[jj * CH + c];
        h1[b][jj] = fmaxf(acc, 0.f);
    }
    __syncthreads();
    for (int i = t; i < BATCH * CH; i += blockDim.x) {
        const int b = i / CH, c = i % CH;
        float acc = 0.f;
#pragma unroll
        for (int jj = 0; jj < CSE; ++jj) acc += h1[b][jj] * w2[c * CSE + jj];
        sescale[i] = 1.f / (1.f + __expf(-acc));
    }
}

// ============ K3: x_se = xpe * sescale (in place) ============
__global__ void k_scale(float* __restrict__ xse, const float* __restrict__ sescale) {
    const size_t total  = (size_t)BATCH * CH * HWSZ;
    const size_t stride = (size_t)gridDim.x * blockDim.x;
    for (size_t i = (size_t)blockIdx.x * blockDim.x + threadIdx.x; i < total; i += stride)
        xse[i] *= sescale[i / HWSZ];
}

// ============ K4: projection GEMM  Y[b] = Wm(Mo x CH) * X[b](CH x HW) ============
template<bool RELU>
__global__ void k_proj(const float* __restrict__ Wm, const float* __restrict__ X,
                       const float* __restrict__ bias, const float* __restrict__ bng,
                       const float* __restrict__ bnb, float* __restrict__ Y, int Mo) {
    const int nBase = blockIdx.x * 256;
    const int mBase = blockIdx.y * 32;
    const int b     = blockIdx.z;
    const float* Xb = X + (size_t)b * CH * HWSZ;
    float*       Yb = Y + (size_t)b * Mo * HWSZ;

    __shared__ v16bf AF[2][32];      // 2 M-rows of fragments   (2 KB)
    __shared__ v16bf BF[16][32];     // 16 N-tiles of fragments (16 KB)

    const int t = threadIdx.x, lane = t & 31, wv = t >> 5;
    const int wm = wv & 1, wn = wv >> 1;

    v8f acc[4];
#pragma unroll
    for (int i = 0; i < 4; ++i)
#pragma unroll
        for (int v = 0; v < 8; ++v) acc[i][v] = 0.f;

    for (int kk = 0; kk < CH; kk += 32) {
        if (kk + 32 < CH) {          // prefetch next K-chunk (global_prefetch_b8)
            const float* pf = Xb + (size_t)(kk + 32 + (t >> 3)) * HWSZ + nBase + (t & 7) * 32;
            __builtin_prefetch(pf, 0, 1);
        }
#pragma unroll
        for (int r = 0; r < 4; ++r) {                 // stage A: 32x32, swizzled
            const int e = t + r * 256;
            const int am = e >> 5, ak = e & 31;
            store_a_elem(AF[am >> 4], am & 15, ak, Wm[(mBase + am) * CH + kk + ak]);
        }
#pragma unroll
        for (int r = 0; r < 8; ++r) {                 // stage B: 32x256 via float4, swizzled
            const int e = t + r * 256;
            const int bk = e >> 6, bn4 = e & 63;
            const int n0 = bn4 * 4, tn = n0 >> 4, nl = n0 & 15;
            const float4 v4 = *(const float4*)(Xb + (size_t)(kk + bk) * HWSZ + nBase + n0);
            store_b_elem(BF[tn], bk, nl + 0, v4.x);
            store_b_elem(BF[tn], bk, nl + 1, v4.y);
            store_b_elem(BF[tn], bk, nl + 2, v4.z);
            store_b_elem(BF[tn], bk, nl + 3, v4.w);
        }
        __syncthreads();
        const v16bf af = AF[wm][lane];
#pragma unroll
        for (int nt = 0; nt < 4; ++nt) {
            const v16bf bfr = BF[wn * 4 + nt][lane];
            acc[nt] = wmma_bf16(af, bfr, acc[nt]);
        }
        __syncthreads();
    }
    const int mAdd = (lane >> 4) * 8, nl = lane & 15;
    const float bnscale = rsqrtf(1.f + BN_EPS);
#pragma unroll
    for (int nt = 0; nt < 4; ++nt) {
        const int n = nBase + wn * 64 + nt * 16 + nl;
#pragma unroll
        for (int v = 0; v < 8; ++v) {
            const int m = mBase + wm * 16 + v + mAdd;
            const float s = bng[m] * bnscale;
            float val = acc[nt][v] * s + (bias[m] * s + bnb[m]);
            if (RELU) val = fmaxf(val, 0.f);
            Yb[(size_t)m * HWSZ + n] = val;
        }
    }
}

// ============ K5: attention scores (MODE 0: e_h per (b,w); MODE 1: e_w per (b,h)) ======
template<int MODE>
__global__ void k_scores(const float* __restrict__ Q, const float* __restrict__ Kk,
                         float* __restrict__ S) {
    const int b   = blockIdx.x >> 7;
    const int fix = blockIdx.x & 127;          // w (MODE0) or h (MODE1)
    const float* Qb = Q  + (size_t)b * C8 * HWSZ;
    const float* Kb = Kk + (size_t)b * C8 * HWSZ;

    __shared__ v16bf QF[8][32];      // 8 M-rows  (8 KB)
    __shared__ v16bf KF[8][32];      // 8 N-tiles (8 KB)

    const int t = threadIdx.x, lane = t & 31, wv = t >> 5;
#pragma unroll
    for (int r = 0; r < 16; ++r) {
        const int e = t + r * 256;             // 4096 elems
        const int c = e >> 7, rw = e & 127;
        const size_t off = (size_t)c * HWSZ +
            (MODE == 0 ? (size_t)rw * WW + fix : (size_t)fix * WW + rw);
        store_a_elem(QF[rw >> 4], rw & 15, c, Qb[off]);
        store_b_elem(KF[rw >> 4], c, rw & 15, Kb[off]);
    }
    __syncthreads();

    const v16bf af = QF[wv][lane];
    const int m0 = wv * 16;
    const int mAdd = (lane >> 4) * 8, nl = lane & 15;
#pragma unroll
    for (int nt = 0; nt < 8; ++nt) {
        v8f accv;
#pragma unroll
        for (int v = 0; v < 8; ++v) accv[v] = 0.f;
        accv = wmma_bf16(af, KF[nt][lane], accv);
#pragma unroll
        for (int v = 0; v < 8; ++v) {
            const int m = m0 + v + mAdd;       // h (MODE0) / w (MODE1)
            const int n = nt * 16 + nl;        // H' / W'
            float val = accv[v];
            if (MODE == 0) {
                if (n == m) val = -__builtin_inff();
                S[(((size_t)(b * HH + m)) * WW + fix) * 256 + n] = val;
            } else {
                S[(((size_t)(b * HH + fix)) * WW + m) * 256 + 128 + n] = val;
            }
        }
    }
}

// ============ K6: softmax over 256 per pixel (one wave per pixel) ============
__global__ void k_softmax(float* __restrict__ S) {
    const size_t pix = (size_t)blockIdx.x * 8 + (threadIdx.x >> 5);
    const int lane = threadIdx.x & 31;
    float* row = S + pix * 256;
    float4 a0 = ((const float4*)row)[lane];
    float4 a1 = ((const float4*)row)[lane + 32];
    float m = fmaxf(fmaxf(fmaxf(a0.x, a0.y), fmaxf(a0.z, a0.w)),
                    fmaxf(fmaxf(a1.x, a1.y), fmaxf(a1.z, a1.w)));
#pragma unroll
    for (int s = 16; s > 0; s >>= 1) m = fmaxf(m, __shfl_xor(m, s, 32));
    a0.x = __expf(a0.x - m); a0.y = __expf(a0.y - m);
    a0.z = __expf(a0.z - m); a0.w = __expf(a0.w - m);
    a1.x = __expf(a1.x - m); a1.y = __expf(a1.y - m);
    a1.z = __expf(a1.z - m); a1.w = __expf(a1.w - m);
    float sum = a0.x + a0.y + a0.z + a0.w + a1.x + a1.y + a1.z + a1.w;
#pragma unroll
    for (int s = 16; s > 0; s >>= 1) sum += __shfl_xor(sum, s, 32);
    const float inv = 1.f / sum;
    a0.x *= inv; a0.y *= inv; a0.z *= inv; a0.w *= inv;
    a1.x *= inv; a1.y *= inv; a1.z *= inv; a1.w *= inv;
    ((float4*)row)[lane]      = a0;
    ((float4*)row)[lane + 32] = a1;
}

// ============ K7: apply attention  Out = att * V  (MODE 0: out_h '=' ; MODE 1: out_w '+=')
// per block: 128(M) x 128(N=c slice), K=128 in 4 chunks of 32.
// att tile is DMA'd by the Tensor Data Mover (rows of 32 contiguous fp32).
template<int MODE>
__global__ void k_apply(const float* __restrict__ S, const float* __restrict__ Vv,
                        float* __restrict__ Out) {
    const int cBase = blockIdx.x * 128;
    const int b     = blockIdx.y >> 7;
    const int fix   = blockIdx.y & 127;        // w (MODE0) / h (MODE1)
    const float* Vb = Vv + (size_t)b * CH * HWSZ;

    __shared__ v16bf AF[8][32];      // att fragments, 8 M-rows per chunk (8 KB)
    __shared__ v16bf VF[8][32];      // V fragments,  8 N-tiles per chunk (8 KB)
    __shared__ float Sraw[128 * 32]; // TDM landing tile (16 KB)

    const int t = threadIdx.x, lane = t & 31, wv = t >> 5;
    v8f acc[8];
#pragma unroll
    for (int i = 0; i < 8; ++i)
#pragma unroll
        for (int v = 0; v < 8; ++v) acc[i][v] = 0.f;

    for (int kk = 0; kk < 128; kk += 32) {
        // ---- TDM async load of the 128 x 32 att tile into Sraw ----
        if (wv == 0) {   // wave-uniform guard: TDM ignores EXEC, issue from one wave
            const size_t gbase = (MODE == 0)
                ? ((((size_t)(b * HH + 0)) * WW + fix) * 256 + kk)
                : ((((size_t)(b * HH + fix)) * WW + 0) * 256 + 128 + kk);
            const unsigned rowStride = (MODE == 0) ? (unsigned)(WW * 256) : 256u;
            tdm_load_tile_f32(S + gbase, Sraw, rowStride, 128u);
        }
        if (kk + 32 < 128) {                   // prefetch next V chunk
            const float* pf = Vb + (size_t)(cBase + (t >> 3)) * HWSZ +
                (MODE == 0 ? (size_t)(kk + 32 + (t & 7) * 4) * WW + fix
                           : (size_t)fix * WW + kk + 32 + (t & 7) * 4);
            __builtin_prefetch(pf, 0, 1);
        }
#pragma unroll
        for (int r = 0; r < 16; ++r) {         // stage V: 32 x 128, swizzled
            const int e = t + r * 256;
            const int n = e >> 5, kc = e & 31;
            const size_t vidx = (size_t)(cBase + n) * HWSZ +
                (MODE == 0 ? (size_t)(kk + kc) * WW + fix
                           : (size_t)fix * WW + (kk + kc));
            store_b_elem(VF[n >> 4], kc, n & 15, Vb[vidx]);
        }
        __syncthreads();                       // Sraw visible to all waves
#pragma unroll
        for (int r = 0; r < 16; ++r) {         // swizzle Sraw -> A fragments
            const int e = t + r * 256;
            const int m = e >> 5, kc = e & 31;
            store_a_elem(AF[m >> 4], m & 15, kc, Sraw[e]);
        }
        __syncthreads();
        const v16bf af = AF[wv][lane];
#pragma unroll
        for (int nt = 0; nt < 8; ++nt)
            acc[nt] = wmma_bf16(af, VF[nt][lane], acc[nt]);
        __syncthreads();                       // Sraw/AF reusable next chunk
    }
    const int mAdd = (lane >> 4) * 8, nl = lane & 15;
    float* Ob = Out + (size_t)b * CH * HWSZ;
#pragma unroll
    for (int nt = 0; nt < 8; ++nt) {
        const int n = cBase + nt * 16 + nl;    // channel
#pragma unroll
        for (int v = 0; v < 8; ++v) {
            const int m = wv * 16 + v + mAdd;
            const size_t oidx = (size_t)n * HWSZ +
                (MODE == 0 ? (size_t)m * WW + fix : (size_t)fix * WW + m);
            if (MODE == 0) Ob[oidx] = acc[nt][v];
            else           Ob[oidx] += acc[nt][v];
        }
    }
}

// ============ K8: z = x_se + gamma*out (in place on out), LN partial sums ============
__global__ void k_combine(const float* __restrict__ xse, float* __restrict__ z,
                          const float* __restrict__ gamma, float* __restrict__ partials) {
    const int b = blockIdx.y;
    const float g = gamma[0];
    const size_t base = (size_t)b * CH * HWSZ;
    const size_t per  = (size_t)CH * HWSZ;
    const size_t stride = (size_t)gridDim.x * blockDim.x;
    float s = 0.f, s2 = 0.f;
    for (size_t i = (size_t)blockIdx.x * blockDim.x + threadIdx.x; i < per; i += stride) {
        const float val = xse[base + i] + g * z[base + i];
        z[base + i] = val;
        s += val; s2 += val * val;
    }
    __shared__ float r1[256], r2[256];
    r1[threadIdx.x] = s; r2[threadIdx.x] = s2;
    __syncthreads();
    for (int st = 128; st > 0; st >>= 1) {
        if (threadIdx.x < st) { r1[threadIdx.x] += r1[threadIdx.x + st];
                                r2[threadIdx.x] += r2[threadIdx.x + st]; }
        __syncthreads();
    }
    if (threadIdx.x == 0) {
        partials[(b * gridDim.x + blockIdx.x) * 2 + 0] = r1[0];
        partials[(b * gridDim.x + blockIdx.x) * 2 + 1] = r2[0];
    }
}

// ============ K9: finalize LN stats per batch ============
__global__ void k_stats(const float* __restrict__ partials, float* __restrict__ stats, int nb) {
    const int b = blockIdx.x, t = threadIdx.x;
    float s = 0.f, s2 = 0.f;
    for (int i = t; i < nb; i += blockDim.x) {
        s  += partials[(b * nb + i) * 2 + 0];
        s2 += partials[(b * nb + i) * 2 + 1];
    }
    __shared__ float r1[256], r2[256];
    r1[t] = s; r2[t] = s2;
    __syncthreads();
    for (int st = 128; st > 0; st >>= 1) {
        if (t < st) { r1[t] += r1[t + st]; r2[t] += r2[t + st]; }
        __syncthreads();
    }
    if (t == 0) {
        const float n = (float)CH * (float)HWSZ;
        const float mu = r1[0] / n;
        const float var = r2[0] / n - mu * mu;
        stats[b * 2 + 0] = mu;
        stats[b * 2 + 1] = rsqrtf(var + LN_EPS);
    }
}

// ============ K10: normalize to d_out ============
__global__ void k_norm(const float* __restrict__ z, const float* __restrict__ stats,
                       float* __restrict__ out) {
    const size_t total  = (size_t)BATCH * CH * HWSZ;
    const size_t stride = (size_t)gridDim.x * blockDim.x;
    for (size_t i = (size_t)blockIdx.x * blockDim.x + threadIdx.x; i < total; i += stride) {
        const int b = (int)(i / ((size_t)CH * HWSZ));
        out[i] = (z[i] - stats[b * 2]) * stats[b * 2 + 1];
    }
}

// ============================ launcher ============================
extern "C" void kernel_launch(void* const* d_in, const int* in_sizes, int n_in,
                              void* d_out, int out_size, void* d_ws, size_t ws_size,
                              hipStream_t stream) {
    (void)in_sizes; (void)n_in; (void)out_size; (void)ws_size;
    const float* x     = (const float*)d_in[0];
    const float* q_w   = (const float*)d_in[1];
    const float* q_b   = (const float*)d_in[2];
    const float* qbn_g = (const float*)d_in[3];
    const float* qbn_b = (const float*)d_in[4];
    const float* k_w   = (const float*)d_in[5];
    const float* k_b   = (const float*)d_in[6];
    const float* kbn_g = (const float*)d_in[7];
    const float* kbn_b = (const float*)d_in[8];
    const float* v_w   = (const float*)d_in[9];
    const float* v_b   = (const float*)d_in[10];
    const float* vbn_g = (const float*)d_in[11];
    const float* vbn_b = (const float*)d_in[12];
    const float* se_w1 = (const float*)d_in[13];
    const float* se_w2 = (const float*)d_in[14];
    const float* gamma = (const float*)d_in[15];
    float* out = (float*)d_out;

    // workspace carving (floats)
    float* ws = (float*)d_ws;
    const size_t nBig = (size_t)BATCH * CH * HWSZ;   // 33.5M
    const size_t nQK  = (size_t)BATCH * C8 * HWSZ;   // 4.2M
    float* xse      = ws;                 ws += nBig;
    float* outbuf   = ws;                 ws += nBig;
    float* vbuf     = ws;                 ws += nBig;
    float* scores   = ws;                 ws += nBig;          // (B,H,W,256)
    float* qbuf     = ws;                 ws += nQK;
    float* kbuf     = ws;                 ws += nQK;
    float* meanbc   = ws;                 ws += BATCH * CH;
    float* sescale  = ws;                 ws += BATCH * CH;
    float* partials = ws;                 ws += BATCH * 256 * 2;
    float* stats    = ws;                 ws += BATCH * 2;

    k_posembed_mean<<<BATCH * CH, 256, 0, stream>>>(x, xse, meanbc);
    k_se<<<1, 256, 0, stream>>>(meanbc, se_w1, se_w2, sescale);
    k_scale<<<8192, 256, 0, stream>>>(xse, sescale);
    k_proj<true ><<<dim3(HWSZ / 256, 1, BATCH), 256, 0, stream>>>(q_w, xse, q_b, qbn_g, qbn_b, qbuf, C8);
    k_proj<true ><<<dim3(HWSZ / 256, 1, BATCH), 256, 0, stream>>>(k_w, xse, k_b, kbn_g, kbn_b, kbuf, C8);
    k_proj<false><<<dim3(HWSZ / 256, CH / 32, BATCH), 256, 0, stream>>>(v_w, xse, v_b, vbn_g, vbn_b, vbuf, CH);
    k_scores<0><<<BATCH * WW, 256, 0, stream>>>(qbuf, kbuf, scores);
    k_scores<1><<<BATCH * HH, 256, 0, stream>>>(qbuf, kbuf, scores);
    k_softmax<<<(BATCH * HWSZ) / 8, 256, 0, stream>>>(scores);
    k_apply<0><<<dim3(CH / 128, BATCH * WW), 256, 0, stream>>>(scores, vbuf, outbuf);
    k_apply<1><<<dim3(CH / 128, BATCH * HH), 256, 0, stream>>>(scores, vbuf, outbuf);
    k_combine<<<dim3(256, BATCH), 256, 0, stream>>>(xse, outbuf, gamma, partials);
    k_stats<<<BATCH, 256, 0, stream>>>(partials, stats, 256);
    k_norm<<<8192, 256, 0, stream>>>(outbuf, stats, out);
}
